// get_aligned_feat_5153960755428
// MI455X (gfx1250) — matrix operand order repaired and checked by reading, not compiled, not run on hardware
//
#include <hip/hip_runtime.h>
#include <hip/hip_bf16.h>
#include <math.h>

typedef __bf16 v16bf __attribute__((ext_vector_type(16)));
typedef __bf16 v8bf  __attribute__((ext_vector_type(8)));
typedef float  v8f   __attribute__((ext_vector_type(8)));

#define BD 192  // 6 waves of 32; each wave owns a 16-wide W segment (6*16 = 96 = W)

static __device__ __forceinline__ v16bf cat16(v8bf lo, v8bf hi) {
  return __builtin_shufflevector(lo, hi, 0, 1, 2, 3, 4, 5, 6, 7,
                                 8, 9, 10, 11, 12, 13, 14, 15);
}

// ---------------------------------------------------------------------------
// concat along channels: cat[b, 0:C] = fr, cat[b, C:2C] = ft
// ---------------------------------------------------------------------------
__global__ void concat_kernel(const float* __restrict__ fr,
                              const float* __restrict__ ft,
                              float* __restrict__ cat,
                              int HW, int C, int n) {
  int i = blockIdx.x * blockDim.x + threadIdx.x;
  if (i >= n) return;
  int rest = i % HW;
  int c = (i / HW) % (2 * C);
  int b = i / (HW * 2 * C);
  cat[i] = (c < C) ? fr[(b * C + c) * HW + rest]
                   : ft[(b * C + (c - C)) * HW + rest];
}

// ---------------------------------------------------------------------------
// Dense 3x3 conv, stride 1, pad 1, NCHW fp32, implicit GEMM via bf16 WMMA.
// K ordering: k = tap*inC + c (inC multiple of 32 -> a 32-chunk stays in one
// tap). LDS activation staging is channel-innermost so each lane's B fragment
// is 16 contiguous bf16 -> 2x ds_load_b128, no element packing.
// ---------------------------------------------------------------------------
__global__ __launch_bounds__(BD) void conv3x3_wmma(
    const float* __restrict__ x, const float* __restrict__ wgt,
    float* __restrict__ out, int H, int W, int inC, int outC, int MT) {
  __shared__ alignas(64) __bf16 sW[16][1152];   // 16-row M tile (max inC=128)
  __shared__ alignas(64) __bf16 sXT[3][98][32]; // [row][padded col][channel]

  int bid = blockIdx.x;
  int mt = bid % MT;
  int h  = (bid / MT) % H;
  int b  = bid / (MT * H);

  int tid  = threadIdx.x;
  int lane = tid & 31;
  int wv   = tid >> 5;
  int w0   = wv << 4;
  int n    = lane & 15;   // N column
  int hf   = lane >> 4;   // lane half
  int m    = lane & 15;   // M row for A fragment

  int K = inC * 9;
  // stage weights: sW[m][tap*inC + c] = wgt[oc, c, ky, kx]
  for (int idx = tid; idx < 16 * K; idx += BD) {
    int mm  = idx / K;
    int k   = idx % K;
    int tap = k / inC;
    int c   = k % inC;
    int oc  = mt * 16 + mm;
    float v = (oc < outC) ? wgt[(oc * inC + c) * 9 + tap] : 0.0f;
    sW[mm][k] = (__bf16)v;
  }
  __syncthreads();

  v8f acc = {};
  for (int c0 = 0; c0 < inC; c0 += 32) {
    // stage 32 channels x 3 rows, zero-padded borders, channel-innermost
    for (int idx = tid; idx < 32 * 3 * 98; idx += BD) {
      int cc  = idx / (3 * 98);
      int rem = idx % (3 * 98);
      int r   = rem / 98;
      int col = rem % 98;
      int wc  = col - 1;
      int hh  = h - 1 + r;
      float v = 0.0f;
      if (hh >= 0 && hh < H && wc >= 0 && wc < W)
        v = x[((b * inC + c0 + cc) * H + hh) * W + wc];
      sXT[r][col][cc] = (__bf16)v;
    }
    __syncthreads();

#pragma unroll
    for (int tap = 0; tap < 9; ++tap) {
      int ky = tap / 3, kx = tap % 3;
      // A (this code's contiguous k order, same as B): K = 16*half + e
      const __bf16* wrow = &sW[m][tap * inC + c0 + (hf << 4)];
      v8bf alo = *(const v8bf*)(wrow);
      v8bf ahi = *(const v8bf*)(wrow + 8);
      v16bf a = cat16(alo, ahi);
      // B (same contiguous k order as A): K = 16*half + e = channel in chunk
      v16bf bm = *(const v16bf*)(&sXT[ky][w0 + n + kx][hf << 4]);
      acc = __builtin_amdgcn_wmma_f32_16x16x32_bf16(false, a, false, bm,
                                                    (short)0, acc, false, false);
    }
    __syncthreads();
  }

  // D layout (f32 16x16): VGPR r -> M = r + 8*half, N = lane%16
#pragma unroll
  for (int r = 0; r < 8; ++r) {
    int oc = mt * 16 + r + (hf << 3);
    if (oc < outC)
      out[((b * outC + oc) * H + h) * W + w0 + n] = acc[r];
  }
}

// ---------------------------------------------------------------------------
// Deformable grouped 3x3 conv (torchvision semantics), bf16 WMMA.
// K ordering kk = tap*Cg + c  =>  for a 32-chunk the tap is uniform per lane:
//   t = (32*ch + 16*half) / Cg   (Cg in {16,32}, templated -> full unroll,
//   all per-chunk sample params live in registers, no dynamic indexing)
// Bilinear sampling is branchless: validity folded into corner weights,
// indices clamped. Weights staged zero-padded in LDS -> unguarded A loads.
// ---------------------------------------------------------------------------
template <int Cg>
__global__ __launch_bounds__(BD) void deform_conv_wmma(
    const float* __restrict__ x, const float* __restrict__ off,
    const float* __restrict__ wgt, const float* __restrict__ bias,
    float* __restrict__ out, int H, int W, int G, int outCg, int outC) {
  constexpr int NK  = 9 * Cg;
  constexpr int NCH = (NK + 31) / 32;
  constexpr int NKP = NCH * 32;
  __shared__ alignas(64) __bf16 sWd[16][NKP];

  const int C = Cg * G;
  int MT = outCg >> 4;
  int bid = blockIdx.x;
  int mt = bid % MT;
  int h  = (bid / MT) % H;
  int g  = (bid / (MT * H)) % G;
  int b  = bid / (MT * H * G);

  int tid  = threadIdx.x;
  int lane = tid & 31;
  int wv   = tid >> 5;
  int w0   = wv << 4;
  int n    = lane & 15;
  int hf   = lane >> 4;
  int m    = lane & 15;
  int w    = w0 + n;
  int offC = G * 18;
  const size_t HW = (size_t)H * W;

  // stage grouped weights, K-order kk = tap*Cg + c, zero padding to NKP
  for (int idx = tid; idx < 16 * NKP; idx += BD) {
    int mm = idx / NKP;
    int kk = idx % NKP;
    float v = 0.0f;
    if (kk < NK) {
      int k = kk / Cg, c = kk % Cg;
      int oc = g * outCg + mt * 16 + mm;
      v = wgt[(oc * Cg + c) * 9 + k];
    }
    sWd[mm][kk] = (__bf16)v;
  }
  __syncthreads();

  // per-lane, per-chunk branchless bilinear params (registers, unrolled index)
  int   i00[NCH], i01[NCH], i10[NCH], i11[NCH];
  float c00[NCH], c01[NCH], c10[NCH], c11[NCH];
#pragma unroll
  for (int ch = 0; ch < NCH; ++ch) {
    int t = (32 * ch + 16 * hf) / Cg;  // tap for this lane's chunk
    i00[ch] = i01[ch] = i10[ch] = i11[ch] = 0;
    c00[ch] = c01[ch] = c10[ch] = c11[ch] = 0.0f;
    if (t < 9) {
      int ky = t / 3, kx = t % 3;
      size_t ob = ((size_t)(b * offC + g * 18 + t * 2) * H + h) * W + w;
      float py = off[ob] + (float)(h - 1 + ky);
      float px = off[ob + HW] + (float)(w - 1 + kx);
      float y0f = floorf(py);
      float x0f = floorf(px);
      int y0 = (int)y0f, x0 = (int)x0f;
      float wy = py - y0f, wx = px - x0f;
      float vy0 = (y0 >= 0 && y0 < H) ? 1.0f : 0.0f;
      float vy1 = (y0 + 1 >= 0 && y0 + 1 < H) ? 1.0f : 0.0f;
      float vx0 = (x0 >= 0 && x0 < W) ? 1.0f : 0.0f;
      float vx1 = (x0 + 1 >= 0 && x0 + 1 < W) ? 1.0f : 0.0f;
      int y0c = min(max(y0, 0), H - 1);
      int y1c = min(max(y0 + 1, 0), H - 1);
      int x0c = min(max(x0, 0), W - 1);
      int x1c = min(max(x0 + 1, 0), W - 1);
      i00[ch] = y0c * W + x0c;  c00[ch] = (1.f - wy) * (1.f - wx) * vy0 * vx0;
      i01[ch] = y0c * W + x1c;  c01[ch] = (1.f - wy) * wx         * vy0 * vx1;
      i10[ch] = y1c * W + x0c;  c10[ch] = wy * (1.f - wx)         * vy1 * vx0;
      i11[ch] = y1c * W + x1c;  c11[ch] = wy * wx                 * vy1 * vx1;
    }
  }

  const float* xg = x + (size_t)(b * C + g * Cg) * HW;  // group base plane
  v8f acc = {};
#pragma unroll
  for (int ch = 0; ch < NCH; ++ch) {
    int kk0 = ch << 5;
    // A fragment: two contiguous aligned 16B LDS loads (zero-padded rows)
    const __bf16* wrow = &sWd[m][kk0 + (hf << 4)];
    v8bf alo = *(const v8bf*)(wrow);
    v8bf ahi = *(const v8bf*)(wrow + 8);
    v16bf a = cat16(alo, ahi);
    // B fragment: channel base within chunk for this lane half
    int cBase = (kk0 + 16 * hf) % Cg;  // compile-time-foldable per ch,Cg
    v16bf bm;
#pragma unroll
    for (int e = 0; e < 16; ++e) {
      const float* xp = xg + (size_t)(cBase + e) * HW;
      float bv = c00[ch] * xp[i00[ch]] + c01[ch] * xp[i01[ch]]
               + c10[ch] * xp[i10[ch]] + c11[ch] * xp[i11[ch]];
      bm[e] = (__bf16)bv;
    }
    acc = __builtin_amdgcn_wmma_f32_16x16x32_bf16(false, a, false, bm,
                                                  (short)0, acc, false, false);
  }

#pragma unroll
  for (int r = 0; r < 8; ++r) {
    int oc = g * outCg + mt * 16 + r + (hf << 3);
    out[((size_t)(b * outC + oc) * H + h) * W + w0 + n] = acc[r] + bias[oc];
  }
}

// ---------------------------------------------------------------------------
extern "C" void kernel_launch(void* const* d_in, const int* in_sizes, int n_in,
                              void* d_out, int out_size, void* d_ws, size_t ws_size,
                              hipStream_t stream) {
  (void)in_sizes; (void)n_in; (void)out_size; (void)ws_size;
  const float* fr     = (const float*)d_in[0];
  const float* ft     = (const float*)d_in[1];
  const float* off1_w = (const float*)d_in[2];
  const float* def1_w = (const float*)d_in[3];
  const float* def1_b = (const float*)d_in[4];
  const float* off2_w = (const float*)d_in[5];
  const float* def2_w = (const float*)d_in[6];
  const float* def2_b = (const float*)d_in[7];
  const float* off3_w = (const float*)d_in[8];
  const float* def3_w = (const float*)d_in[9];
  const float* def3_b = (const float*)d_in[10];
  const float* off4_w = (const float*)d_in[11];
  const float* def4_w = (const float*)d_in[12];
  const float* def4_b = (const float*)d_in[13];

  const int B = 8, C = 64, H = 96, W = 96;
  const size_t HW = (size_t)H * W;

  float* ws    = (float*)d_ws;
  float* cat   = ws;                                  // [B,128,H,W]
  float* offb  = cat   + (size_t)B * 128 * HW;        // [B,72,H,W] (>= 36)
  float* feat1 = offb  + (size_t)B * 72  * HW;        // [B,64,H,W]
  float* feat3 = feat1 + (size_t)B * 64  * HW;        // [B,64,H,W]
  float* feat2 = cat;                                 // alias: cat dead after stage 1

  float* outp = (float*)d_out;

  // stage 0: concat
  {
    int nTot = (int)(B * 128 * HW);
    concat_kernel<<<(nTot + 255) / 256, 256, 0, stream>>>(fr, ft, cat, (int)HW, C, nTot);
  }
  // stage 1: offsets from cat (inC=128, outC=72), deform on cat (G=4, Cg=32)
  conv3x3_wmma<<<B * H * 5, BD, 0, stream>>>(cat, off1_w, offb, H, W, 128, 72, 5);
  deform_conv_wmma<32><<<B * 4 * H * 1, BD, 0, stream>>>(cat, offb, def1_w, def1_b,
                                                         feat1, H, W, 4, 16, 64);
  // stage 2 (G=4, Cg=16)
  conv3x3_wmma<<<B * H * 5, BD, 0, stream>>>(feat1, off2_w, offb, H, W, 64, 72, 5);
  deform_conv_wmma<16><<<B * 4 * H * 1, BD, 0, stream>>>(feat1, offb, def2_w, def2_b,
                                                         feat2, H, W, 4, 16, 64);
  // stage 3 (G=4, Cg=16)
  conv3x3_wmma<<<B * H * 5, BD, 0, stream>>>(feat2, off3_w, offb, H, W, 64, 72, 5);
  deform_conv_wmma<16><<<B * 4 * H * 1, BD, 0, stream>>>(feat2, offb, def3_w, def3_b,
                                                         feat3, H, W, 4, 16, 64);
  // stage 4: offsets from feat3 (outC=36), deform applied to ft (G=2, Cg=32)
  conv3x3_wmma<<<B * H * 3, BD, 0, stream>>>(feat3, off4_w, offb, H, W, 64, 36, 3);
  deform_conv_wmma<32><<<B * 2 * H * 2, BD, 0, stream>>>(ft, offb, def4_w, def4_b,
                                                         outp, H, W, 2, 32, 64);
}
